// SPO_Model_Simple_19602230739597
// MI455X (gfx1250) — matrix operand order repaired and checked by reading, not compiled, not run on hardware
//
#include <hip/hip_runtime.h>
#include <hip/hip_bf16.h>

// ---------------------------------------------------------------------------
// BiLSTM + MLP head + CRF Viterbi for MI455X (gfx1250, wave32, WMMA)
// B=64 T=256 V=50000 E=300 H=512 HID=100 K=9
// ---------------------------------------------------------------------------

#define Bb 64
#define Tt 256
#define Ee 300
#define EeP 320          // E padded to multiple of 64 for pipelined bf16 WMMA
#define Hh 512
#define G4 2048          // 4*H
#define BT (Bb * Tt)     // 16384
#define HIDp 128         // 100 padded to 128 (2 N-tiles of 64)

typedef __attribute__((ext_vector_type(16))) __bf16 v16bf;
typedef __attribute__((ext_vector_type(8)))  float  v8f;

union BF16x16 { uint4 q[2]; v16bf v; };

__device__ __forceinline__ float sigf(float x) { return 1.0f / (1.0f + __expf(-x)); }

// ---------------------------------------------------------------------------
// init: zero h state double-buffers and grid-barrier words
// ---------------------------------------------------------------------------
__global__ void init_kernel(__bf16* hstate, unsigned* bar) {
    int idx = blockIdx.x * 256 + threadIdx.x;
    if (idx < 2 * 2 * Bb * Hh) hstate[idx] = (__bf16)0.0f;
    if (idx < 4) bar[idx] = 0u;
}

// ---------------------------------------------------------------------------
// embedding gather + fp32 -> bf16, pad E 300 -> 320 with zeros
// ---------------------------------------------------------------------------
__global__ void embed_kernel(const int* __restrict__ X, const float* __restrict__ emb,
                             __bf16* __restrict__ xbf) {
    int idx = blockIdx.x * 256 + threadIdx.x;
    if (idx >= BT * EeP) return;
    int row = idx / EeP;
    int k   = idx % EeP;
    float v = 0.0f;
    if (k < Ee) {
        int tok = X[row];
        v = emb[(size_t)tok * Ee + k];
    }
    xbf[idx] = (__bf16)v;
}

// fp32 (rIn x kIn) -> bf16 (rOut x kOut), zero padded
__global__ void conv_pad_kernel(const float* __restrict__ src, __bf16* __restrict__ dst,
                                int rIn, int rOut, int kIn, int kOut) {
    int idx = blockIdx.x * 256 + threadIdx.x;
    if (idx >= rOut * kOut) return;
    int r = idx / kOut, k = idx % kOut;
    float v = (r < rIn && k < kIn) ? src[(size_t)r * kIn + k] : 0.0f;
    dst[idx] = (__bf16)v;
}

__global__ void add_bias_kernel(const float* a, const float* b, float* d, int n) {
    int i = blockIdx.x * 256 + threadIdx.x;
    if (i < n) d[i] = a[i] + b[i];
}

__global__ void pad_bias_kernel(const float* s, float* d, int nIn, int nOut) {
    int i = blockIdx.x * 256 + threadIdx.x;
    if (i < nOut) d[i] = (i < nIn) ? s[i] : 0.0f;
}

// ---------------------------------------------------------------------------
// Generic bf16 WMMA GEMM:  C[M,N] = A[M,K] * B[N,K]^T + bias[N]   (f32 accum)
// block = 256 threads = 8 waves, tile 128x64 (each wave: 16 rows x 64 cols)
// K must be a multiple of 64; all rows 16B aligned.
// Software pipeline with two STATIC buffer sets (no runtime indexing ->
// everything stays in registers, no scratch).
// ---------------------------------------------------------------------------
__global__ __launch_bounds__(256) void gemm_bf16_wmma(
    const __bf16* __restrict__ A, const __bf16* __restrict__ Bw,
    const float* __restrict__ bias, float* __restrict__ C,
    int M, int N, int K) {
    const int lane = threadIdx.x & 31;
    const int wave = threadIdx.x >> 5;        // 0..7
    const int half = lane >> 4;               // 0 or 1
    const int l16  = lane & 15;
    const int m0 = blockIdx.y * 128 + wave * 16;
    const int n0 = blockIdx.x * 64;

    const __bf16* arow  = A  + (size_t)(m0 + l16) * K + half * 8;
    const __bf16* brow0 = Bw + (size_t)(n0 + l16) * K + half * 8;

    v8f acc[4] = {};
    BF16x16 a0, a1, b0[4], b1[4];

    auto loadK = [&](int kb, BF16x16& a, BF16x16 (&b)[4]) {
        const __bf16* ap = arow + kb;
        a.q[0] = *(const uint4*)(ap);
        a.q[1] = *(const uint4*)(ap + 16);
        #pragma unroll
        for (int j = 0; j < 4; ++j) {
            const __bf16* bp = brow0 + (size_t)j * 16 * K + kb;
            b[j].q[0] = *(const uint4*)(bp);
            b[j].q[1] = *(const uint4*)(bp + 16);
        }
    };
    auto mm = [&](BF16x16& a, BF16x16 (&b)[4]) {
        #pragma unroll
        for (int j = 0; j < 4; ++j) {
            acc[j] = __builtin_amdgcn_wmma_f32_16x16x32_bf16(
                false, a.v, false, b[j].v, (short)0, acc[j], false, false);
        }
    };

    loadK(0, a0, b0);
    for (int kb = 0; kb < K; kb += 64) {
        loadK(kb + 32, a1, b1);
        if (kb + 96 < K) __builtin_prefetch(arow + kb + 96, 0, 3);
        mm(a0, b0);
        if (kb + 64 < K) loadK(kb + 64, a0, b0);
        mm(a1, b1);
    }

    #pragma unroll
    for (int j = 0; j < 4; ++j) {
        int n = n0 + j * 16 + l16;
        float bv = bias[n];
        #pragma unroll
        for (int v = 0; v < 8; ++v) {
            int m = m0 + v + half * 8;
            C[(size_t)m * N + n] = acc[j][v] + bv;
        }
    }
}

// ---------------------------------------------------------------------------
// sense-reversal grid barrier across the 16 blocks of one direction
// ---------------------------------------------------------------------------
__device__ void grid_barrier(unsigned* cnt, unsigned* phase, unsigned nb) {
    __syncthreads();
    if (threadIdx.x == 0) {
        __threadfence();
        unsigned ph = __hip_atomic_load(phase, __ATOMIC_ACQUIRE, __HIP_MEMORY_SCOPE_AGENT);
        unsigned a  = __hip_atomic_fetch_add(cnt, 1u, __ATOMIC_ACQ_REL, __HIP_MEMORY_SCOPE_AGENT);
        if (a == nb - 1) {
            __hip_atomic_store(cnt, 0u, __ATOMIC_RELAXED, __HIP_MEMORY_SCOPE_AGENT);
            __hip_atomic_fetch_add(phase, 1u, __ATOMIC_ACQ_REL, __HIP_MEMORY_SCOPE_AGENT);
        } else {
            while (__hip_atomic_load(phase, __ATOMIC_ACQUIRE, __HIP_MEMORY_SCOPE_AGENT) == ph)
                __builtin_amdgcn_s_sleep(2);
        }
    }
    __syncthreads();
}

// ---------------------------------------------------------------------------
// Persistent bidirectional LSTM recurrence.
//   grid = 32 blocks: dir = bx/16, wg = bx%16 owns hidden units [wg*32, wg*32+32)
//   Whh slice (4 gates x 32 rows x 512) = 128KB bf16 pinned in LDS whole kernel.
//   h state double-buffered in global (64KB/dir, L2-hot); grid barrier per step.
//   Per step per block: 64x128 gate slice via v_wmma_f32_16x16x32_bf16,
//   k-loop software-pipelined with static double buffers.
// ---------------------------------------------------------------------------
__global__ __launch_bounds__(256) void lstm_rec_kernel(
    const __bf16* __restrict__ WhhF, const __bf16* __restrict__ WhhB,
    const float* __restrict__ XWf, const float* __restrict__ XWb,
    const int* __restrict__ length,
    __bf16* __restrict__ hstate,      // [2 dir][2 buf][64][512] bf16
    __bf16* __restrict__ hcat,        // [B*T][1024] bf16 (hf | hb)
    unsigned* __restrict__ bar) {

    __shared__ __align__(16) __bf16 lW[128 * Hh];   // 128 KB weight slice
    __shared__ __align__(16) float  lG[64 * 128];   //  32 KB gate staging

    const int dir = blockIdx.x >> 4;
    const int wg  = blockIdx.x & 15;
    const int n0  = wg * 32;
    const int tid = threadIdx.x;

    const __bf16* Whh = dir ? WhhB : WhhF;
    const float*  XW  = dir ? XWb  : XWf;
    unsigned* cnt = bar + dir * 2;
    unsigned* ph  = bar + dir * 2 + 1;

    // load this block's Whh slice into LDS once (local row r: gate=r>>5, hid=n0+(r&31))
    for (int i = tid; i < 128 * Hh / 8; i += 256) {      // 16B chunks
        int r  = i >> 6;
        int kk = (i & 63) * 8;
        int grow = (r >> 5) * Hh + n0 + (r & 31);
        *(uint4*)(lW + r * Hh + kk) = *(const uint4*)(Whh + (size_t)grow * Hh + kk);
    }
    __syncthreads();

    const int lane = tid & 31;
    const int wave = tid >> 5;
    const int half = lane >> 4;
    const int l16  = lane & 15;
    const int mt   = wave >> 1;           // M-tile 0..3
    const int ntb  = (wave & 1) * 4;      // N-tile base 0 or 4

    // fixed per-wave LDS base for B fragments
    const __bf16* lwrow = lW + (size_t)(ntb * 16 + l16) * Hh + half * 8;

    const int cm = tid >> 2;              // batch row 0..63 for cell phase
    const int cj = (tid & 3) * 8;         // 8 hidden units per thread
    float creg[8];
    #pragma unroll
    for (int q = 0; q < 8; ++q) creg[q] = 0.0f;
    const int mylen = length[cm];

    for (int step = 0; step < Tt; ++step) {
        const int tt = dir ? (Tt - 1 - step) : step;
        const __bf16* hcur = hstate + (size_t)(dir * 2 + (step & 1)) * (Bb * Hh);
        __bf16*       hnxt = hstate + (size_t)(dir * 2 + ((step + 1) & 1)) * (Bb * Hh);
        const __bf16* arow = hcur + (size_t)(mt * 16 + l16) * Hh + half * 8;

        // ---- gates(64x128) = h_prev(64x512) @ WhhSlice^T via pipelined WMMA ----
        v8f acc[4] = {};
        BF16x16 a0, a1, b0[4], b1[4];
        auto loadK = [&](int kb, BF16x16& a, BF16x16 (&b)[4]) {
            const __bf16* ap = arow + kb;
            a.q[0] = *(const uint4*)(ap);
            a.q[1] = *(const uint4*)(ap + 16);
            #pragma unroll
            for (int j = 0; j < 4; ++j) {
                const __bf16* bp = lwrow + (size_t)j * 16 * Hh + kb;
                b[j].q[0] = *(const uint4*)(bp);
                b[j].q[1] = *(const uint4*)(bp + 16);
            }
        };
        auto mm = [&](BF16x16& a, BF16x16 (&b)[4]) {
            #pragma unroll
            for (int j = 0; j < 4; ++j) {
                acc[j] = __builtin_amdgcn_wmma_f32_16x16x32_bf16(
                    false, a.v, false, b[j].v, (short)0, acc[j], false, false);
            }
        };

        loadK(0, a0, b0);
        #pragma unroll 2
        for (int kb = 0; kb < Hh; kb += 64) {
            loadK(kb + 32, a1, b1);
            mm(a0, b0);
            if (kb + 64 < Hh) loadK(kb + 64, a0, b0);
            mm(a1, b1);
        }

        #pragma unroll
        for (int j = 0; j < 4; ++j) {
            int ncol = (ntb + j) * 16 + l16;
            #pragma unroll
            for (int v = 0; v < 8; ++v) {
                int m = mt * 16 + v + half * 8;
                lG[m * 128 + ncol] = acc[j][v];
            }
        }
        __syncthreads();

        // ---- LSTM cell for this thread's 8 (batch=cm, hid=n0+cj..+7) ----
        const size_t xrow = (size_t)(cm * Tt + tt) * G4;
        const bool msk = tt < mylen;
        #pragma unroll
        for (int q = 0; q < 8; ++q) {
            int j = cj + q;
            int hidg = n0 + j;
            float ig = lG[cm * 128 + j]      + XW[xrow + hidg];
            float fg = lG[cm * 128 + 32 + j] + XW[xrow + 512 + hidg];
            float gg = lG[cm * 128 + 64 + j] + XW[xrow + 1024 + hidg];
            float og = lG[cm * 128 + 96 + j] + XW[xrow + 1536 + hidg];
            float cn = sigf(fg) * creg[q] + sigf(ig) * tanhf(gg);
            float hn = sigf(og) * tanhf(cn);
            float hold = (float)hcur[cm * Hh + hidg];
            float hcarry = msk ? hn : hold;
            if (msk) creg[q] = cn;
            hnxt[cm * Hh + hidg] = (__bf16)hcarry;
            float outv = msk ? hn : 0.0f;
            hcat[(size_t)(cm * Tt + tt) * 1024 + dir * 512 + hidg] = (__bf16)outv;
        }
        grid_barrier(cnt, ph, 16);
    }
}

// ---------------------------------------------------------------------------
// logits = hid[:, :100] @ Wn^T + bn      (tiny: K=9, HID=100)
// ---------------------------------------------------------------------------
__global__ void head_kernel(const float* __restrict__ hid, const float* __restrict__ Wn,
                            const float* __restrict__ bn, float* __restrict__ logits) {
    int idx = blockIdx.x * 128 + threadIdx.x;
    if (idx >= BT * 9) return;
    int row = idx / 9, k = idx % 9;
    const float* hp = hid + (size_t)row * HIDp;
    const float* wp = Wn + k * 100;
    float s = bn[k];
    for (int h = 0; h < 100; ++h) s += hp[h] * wp[h];
    logits[idx] = s;
}

// ---------------------------------------------------------------------------
// CRF Viterbi: one thread per sequence (K=9)
// ---------------------------------------------------------------------------
__global__ void viterbi_kernel(const float* __restrict__ logits, const int* __restrict__ length,
                               const float* __restrict__ start_t, const float* __restrict__ end_t,
                               const float* __restrict__ trans,
                               int* __restrict__ bp, int* __restrict__ out) {
    int b = threadIdx.x;
    if (b >= Bb) return;
    const float* lg = logits + (size_t)b * Tt * 9;
    int* mybp = bp + b * (Tt - 1) * 9;
    float s[9];
    for (int k = 0; k < 9; ++k) s[k] = start_t[k] + lg[k];
    const int len = length[b];
    for (int t = 1; t < Tt; ++t) {
        float ns[9];
        for (int j = 0; j < 9; ++j) {
            float best = -1e30f; int arg = 0;
            for (int i = 0; i < 9; ++i) {
                float c = s[i] + trans[i * 9 + j];
                if (c > best) { best = c; arg = i; }
            }
            mybp[(t - 1) * 9 + j] = arg;
            ns[j] = best + lg[t * 9 + j];
        }
        bool m = t < len;
        for (int j = 0; j < 9; ++j) s[j] = m ? ns[j] : s[j];
    }
    float best = -1e30f; int cur = 0;
    for (int k = 0; k < 9; ++k) {
        float f = s[k] + end_t[k];
        if (f > best) { best = f; cur = k; }
    }
    for (int t = Tt - 1; t >= 1; --t) {
        out[b * Tt + t] = cur;
        if (t < len) cur = mybp[(t - 1) * 9 + cur];
    }
    out[b * Tt] = cur;
}

// ---------------------------------------------------------------------------
extern "C" void kernel_launch(void* const* d_in, const int* in_sizes, int n_in,
                              void* d_out, int out_size, void* d_ws, size_t ws_size,
                              hipStream_t stream) {
    const int*   X      = (const int*)d_in[0];
    const int*   length = (const int*)d_in[2];
    const float* emb    = (const float*)d_in[3];
    const float* Wih_f  = (const float*)d_in[4];
    const float* Whh_f  = (const float*)d_in[5];
    const float* bih_f  = (const float*)d_in[6];
    const float* bhh_f  = (const float*)d_in[7];
    const float* Wih_b  = (const float*)d_in[8];
    const float* Whh_b  = (const float*)d_in[9];
    const float* bih_b  = (const float*)d_in[10];
    const float* bhh_b  = (const float*)d_in[11];
    const float* Wh     = (const float*)d_in[12];
    const float* bh     = (const float*)d_in[13];
    const float* Wn     = (const float*)d_in[14];
    const float* bn     = (const float*)d_in[15];
    const float* startt = (const float*)d_in[16];
    const float* endt   = (const float*)d_in[17];
    const float* trans  = (const float*)d_in[18];

    char* w = (char*)d_ws;
    auto alloc = [&](size_t bytes) { char* p = w; w += (bytes + 255) & ~(size_t)255; return p; };

    __bf16* xbf    = (__bf16*)alloc((size_t)BT * EeP * 2);
    __bf16* wihf   = (__bf16*)alloc((size_t)G4 * EeP * 2);
    __bf16* wihb   = (__bf16*)alloc((size_t)G4 * EeP * 2);
    __bf16* whhf   = (__bf16*)alloc((size_t)G4 * Hh * 2);
    __bf16* whhb   = (__bf16*)alloc((size_t)G4 * Hh * 2);
    __bf16* whp    = (__bf16*)alloc((size_t)HIDp * 1024 * 2);
    float*  biasf  = (float*)alloc((size_t)G4 * 4);
    float*  biasb  = (float*)alloc((size_t)G4 * 4);
    float*  bhp    = (float*)alloc((size_t)HIDp * 4);
    float*  xwf    = (float*)alloc((size_t)BT * G4 * 4);
    float*  xwb    = (float*)alloc((size_t)BT * G4 * 4);
    __bf16* hstate = (__bf16*)alloc((size_t)2 * 2 * Bb * Hh * 2);
    __bf16* hcat   = (__bf16*)alloc((size_t)BT * 1024 * 2);
    float*  hid    = (float*)alloc((size_t)BT * HIDp * 4);
    float*  logits = (float*)alloc((size_t)BT * 9 * 4);
    int*    bpbuf  = (int*)alloc((size_t)Bb * (Tt - 1) * 9 * 4);
    unsigned* bar  = (unsigned*)alloc(256);

    // init state + barrier
    init_kernel<<<(2 * 2 * Bb * Hh + 255) / 256, 256, 0, stream>>>(hstate, bar);

    // embedding gather -> bf16
    embed_kernel<<<(BT * EeP + 255) / 256, 256, 0, stream>>>(X, emb, xbf);

    // weight conversions
    conv_pad_kernel<<<(G4 * EeP + 255) / 256, 256, 0, stream>>>(Wih_f, wihf, G4, G4, Ee, EeP);
    conv_pad_kernel<<<(G4 * EeP + 255) / 256, 256, 0, stream>>>(Wih_b, wihb, G4, G4, Ee, EeP);
    conv_pad_kernel<<<(G4 * Hh + 255) / 256, 256, 0, stream>>>(Whh_f, whhf, G4, G4, Hh, Hh);
    conv_pad_kernel<<<(G4 * Hh + 255) / 256, 256, 0, stream>>>(Whh_b, whhb, G4, G4, Hh, Hh);
    conv_pad_kernel<<<(HIDp * 1024 + 255) / 256, 256, 0, stream>>>(Wh, whp, 100, HIDp, 1024, 1024);
    add_bias_kernel<<<(G4 + 255) / 256, 256, 0, stream>>>(bih_f, bhh_f, biasf, G4);
    add_bias_kernel<<<(G4 + 255) / 256, 256, 0, stream>>>(bih_b, bhh_b, biasb, G4);
    pad_bias_kernel<<<(HIDp + 255) / 256, 256, 0, stream>>>(bh, bhp, 100, HIDp);

    // batched input projections: XW = x @ Wih^T + (bih + bhh)
    gemm_bf16_wmma<<<dim3(G4 / 64, BT / 128), 256, 0, stream>>>(xbf, wihf, biasf, xwf, BT, G4, EeP);
    gemm_bf16_wmma<<<dim3(G4 / 64, BT / 128), 256, 0, stream>>>(xbf, wihb, biasb, xwb, BT, G4, EeP);

    // persistent bidirectional recurrence (32 blocks: 16 per direction)
    lstm_rec_kernel<<<32, 256, 0, stream>>>(whhf, whhb, xwf, xwb, length, hstate, hcat, bar);

    // head: hid = [hf|hb] @ Wh^T + bh   (WMMA, N padded to 128)
    gemm_bf16_wmma<<<dim3(HIDp / 64, BT / 128), 256, 0, stream>>>(hcat, whp, bhp, hid, BT, HIDp, 1024);

    // logits = hid @ Wn^T + bn
    head_kernel<<<(BT * 9 + 127) / 128, 128, 0, stream>>>(hid, Wn, bn, logits);

    // CRF Viterbi decode
    viterbi_kernel<<<1, 64, 0, stream>>>(logits, length, startt, endt, trans, bpbuf, (int*)d_out);
}
